// GraphConflator_8624294330996
// MI455X (gfx1250) — matrix operand order repaired and checked by reading, not compiled, not run on hardware
//
#include <hip/hip_runtime.h>
#include <hip/hip_bf16.h>

// CDNA5 / gfx1250 wave32 WMMA types
typedef __attribute__((ext_vector_type(16))) __bf16 v16bf;
typedef __attribute__((ext_vector_type(8)))  float  v8f;

#define WMMA_BF16(a, b, c) \
  __builtin_amdgcn_wmma_f32_16x16x32_bf16(false, (a), false, (b), (short)0, (c), false, false)

// Fragment placement helpers (inverse of ISA 7.12.2 VGPR layouts).
// A-matrix: lane(half,lr) row=lr, kk=(i<8?i:i+8)+8*half
__device__ __forceinline__ void a_frag_slot(int kk, int& half, int& fi) {
  half = (kk >> 3) & 1;
  fi = (kk & 7) + ((kk >> 4) & 1) * 8;
}
// B-matrix: lane(half,lr) col=lr, kk=(i<8?i:i+8)+8*half
__device__ __forceinline__ void b_frag_slot(int kk, int& half, int& fi) {
  half = (kk >> 3) & 1;
  fi = (kk & 7) + ((kk >> 4) & 1) * 8;
}

// ---------------------------------------------------------------------------
// Small elementwise / graph kernels
// ---------------------------------------------------------------------------
__global__ void gc_fill_ones(float* p, int n) {
  int i = blockIdx.x * blockDim.x + threadIdx.x;
  if (i < n) p[i] = 1.0f;  // self-loop contribution to degree
}

__global__ void gc_deg_scatter(const int* __restrict__ dst, int e, float* deg) {
  int i = blockIdx.x * blockDim.x + threadIdx.x;
  if (i < e) atomicAdd(&deg[dst[i]], 1.0f);
}

__global__ void gc_finish_dinv(float* p, int n) {
  int i = blockIdx.x * blockDim.x + threadIdx.x;
  if (i < n) p[i] = rsqrtf(fmaxf(p[i], 1e-12f));
}

// out[i][f] = xw[i][f] * dinv[i]^2 + bias[f]   (self-loop term + bias)
__global__ void gc_agg_init(const float* __restrict__ xw, const float* __restrict__ dinv,
                            const float* __restrict__ bias, float* __restrict__ out,
                            int n, int d) {
  int idx = blockIdx.x * blockDim.x + threadIdx.x;
  if (idx >= n * d) return;
  int i = idx / d, f = idx % d;
  float w = dinv[i];
  out[idx] = xw[idx] * w * w + bias[f];
}

// per (edge, 4 features): out[dst] += xw[src] * dinv[src]*dinv[dst]
__global__ void gc_agg_scatter(const int* __restrict__ src, const int* __restrict__ dst,
                               const float* __restrict__ xw, const float* __restrict__ dinv,
                               float* __restrict__ out, int e, int d) {
  int t = blockIdx.x * blockDim.x + threadIdx.x;
  int per = d >> 2;
  int edge = t / per;
  if (edge >= e) return;
  int f4 = (t % per) << 2;
  int s = src[edge], dd = dst[edge];
  float norm = dinv[s] * dinv[dd];
  const float4 x = *(const float4*)&xw[(size_t)s * d + f4];
  float* o = &out[(size_t)dd * d + f4];
  atomicAdd(o + 0, x.x * norm);
  atomicAdd(o + 1, x.y * norm);
  atomicAdd(o + 2, x.z * norm);
  atomicAdd(o + 3, x.w * norm);
}

__global__ void gc_relu_ip(float* p, int n) {
  int i = blockIdx.x * blockDim.x + threadIdx.x;
  if (i < n) p[i] = fmaxf(p[i], 0.0f);
}

// ---------------------------------------------------------------------------
// Generic GEMM: C[M,N] = A[M,K] @ B[K,N], f32 in/out, bf16 WMMA inner product.
// Block tile 128x64, 8 waves, wave w owns rows [16w,16w+16) x 64 cols.
// LDS tiles are stored in FRAGMENT-MAJOR order: each lane's 16 bf16 fragment
// values are contiguous -> fragment read = one v16bf load (2x ds_load_b128).
// Requires M%128==0, N%64==0, K%32==0.
// ---------------------------------------------------------------------------
__global__ __launch_bounds__(256) void gemm_bf16(const float* __restrict__ A,
                                                 const float* __restrict__ B,
                                                 float* __restrict__ C,
                                                 int M, int N, int K) {
  __shared__ v16bf Af[8][32];  // [wave][lane]
  __shared__ v16bf Bf[4][32];  // [ntile][lane]
  const int wave = threadIdx.x >> 5, lane = threadIdx.x & 31;
  const int half = lane >> 4, lr = lane & 15;
  const int row0 = blockIdx.x * 128;
  const int col0 = blockIdx.y * 64;
  v8f acc[4] = {};
  for (int k0 = 0; k0 < K; k0 += 32) {
    // A tile 128x32 -> fragment-major
    for (int id = threadIdx.x; id < 128 * 32; id += 256) {
      int r = id >> 5, c = id & 31;
      float av = A[(size_t)(row0 + r) * K + k0 + c];
      int ah, fi; a_frag_slot(c, ah, fi);
      ((__bf16*)&Af[r >> 4][ah * 16 + (r & 15)])[fi] = (__bf16)av;
    }
    // B tile 32x64 -> fragment-major
    for (int id = threadIdx.x; id < 32 * 64; id += 256) {
      int kk = id >> 6, n = id & 63;
      float bv = B[(size_t)(k0 + kk) * N + col0 + n];
      int bh, fi; b_frag_slot(kk, bh, fi);
      ((__bf16*)&Bf[n >> 4][bh * 16 + (n & 15)])[fi] = (__bf16)bv;
    }
    __syncthreads();
    v16bf a = Af[wave][lane];
#pragma unroll
    for (int t = 0; t < 4; ++t)
      acc[t] = WMMA_BF16(a, Bf[t][lane], acc[t]);
    __syncthreads();
  }
  // C fragment: VGPR j -> row j + 8*half, lanes -> cols
#pragma unroll
  for (int t = 0; t < 4; ++t)
#pragma unroll
    for (int j = 0; j < 8; ++j) {
      int r = row0 + wave * 16 + j + 8 * half;
      int c = col0 + t * 16 + lr;
      C[(size_t)r * N + c] = acc[t][j];
    }
}

// ---------------------------------------------------------------------------
// Flash attention: O = softmax(leaky_relu(Q @ K^T)) @ V, d = 64, f32 I/O.
// 128 query rows / block (8 waves x 16 rows); 64-key tiles streamed via LDS.
// All LDS buffers fragment-major.
// ---------------------------------------------------------------------------
__global__ __launch_bounds__(256) void flash_attn(const float* __restrict__ Q,
                                                  const float* __restrict__ K,
                                                  const float* __restrict__ V,
                                                  float* __restrict__ O, int n) {
  __shared__ v16bf Kf[4][2][32];  // B of S=Q@K^T: [keytile][kstep][lane]
  __shared__ v16bf Vf[4][2][32];  // B of P@V:     [vtile][kstep][lane]
  __shared__ v16bf Ps[8][2][32];  // A of P@V:     [wave][kstep][lane]
  const int wave = threadIdx.x >> 5, lane = threadIdx.x & 31;
  const int half = lane >> 4, lr = lane & 15;
  const int qrow = blockIdx.x * 128 + wave * 16 + lr;

  // Q fragments held in registers for entire kernel (2 k-steps of 32)
  v16bf qa[2];
#pragma unroll
  for (int ks = 0; ks < 2; ++ks)
#pragma unroll
    for (int i = 0; i < 16; ++i) {
      int kk = (i < 8 ? i : i + 8) + half * 8 + ks * 32;
      qa[ks][i] = (__bf16)Q[(size_t)qrow * 64 + kk];
    }

  float m[8], l[8];
#pragma unroll
  for (int j = 0; j < 8; ++j) { m[j] = -3.0e38f; l[j] = 0.0f; }
  v8f acc_o[4] = {};

  for (int kt = 0; kt < n; kt += 64) {
    // cooperative coalesced load of K/V tile, scatter into fragment slots
    for (int id = threadIdx.x; id < 64 * 64; id += 256) {
      int key = id >> 6, c = id & 63;
      float kv = K[(size_t)(kt + key) * 64 + c];
      float vv = V[(size_t)(kt + key) * 64 + c];
      {  // K: B-matrix of Q@K^T, kdim=c, col=key
        int ks = c >> 5, rem = c & 31, bh, fi;
        b_frag_slot(rem, bh, fi);
        ((__bf16*)&Kf[key >> 4][ks][bh * 16 + (key & 15)])[fi] = (__bf16)kv;
      }
      {  // V: B-matrix of P@V, kdim=key, col=c
        int ks = key >> 5, rem = key & 31, bh, fi;
        b_frag_slot(rem, bh, fi);
        ((__bf16*)&Vf[c >> 4][ks][bh * 16 + (c & 15)])[fi] = (__bf16)vv;
      }
    }
    __syncthreads();

    // S = Q @ K^T over 4 key tiles of 16
    v8f s[4] = {};
#pragma unroll
    for (int ks = 0; ks < 2; ++ks)
#pragma unroll
      for (int t = 0; t < 4; ++t)
        s[t] = WMMA_BF16(qa[ks], Kf[t][ks][lane], s[t]);

    // leaky_relu + per-row tile max (row j+8*half lives in VGPR j across a half)
    float tm[8];
#pragma unroll
    for (int j = 0; j < 8; ++j) tm[j] = -3.0e38f;
#pragma unroll
    for (int t = 0; t < 4; ++t)
#pragma unroll
      for (int j = 0; j < 8; ++j) {
        float x = s[t][j];
        x = x > 0.0f ? x : 0.01f * x;
        s[t][j] = x;
        tm[j] = fmaxf(tm[j], x);
      }
#pragma unroll
    for (int off = 8; off >= 1; off >>= 1)
#pragma unroll
      for (int j = 0; j < 8; ++j)
        tm[j] = fmaxf(tm[j], __shfl_xor(tm[j], off, 32));

    // online softmax rescale
    float scale[8], rs[8];
#pragma unroll
    for (int j = 0; j < 8; ++j) {
      float mn = fmaxf(m[j], tm[j]);
      scale[j] = __expf(m[j] - mn);
      m[j] = mn;
      rs[j] = 0.0f;
    }
#pragma unroll
    for (int t = 0; t < 4; ++t)
#pragma unroll
      for (int j = 0; j < 8; ++j) {
        float p = __expf(s[t][j] - m[j]);
        s[t][j] = p;
        rs[j] += p;
      }
#pragma unroll
    for (int off = 8; off >= 1; off >>= 1)
#pragma unroll
      for (int j = 0; j < 8; ++j)
        rs[j] += __shfl_xor(rs[j], off, 32);
#pragma unroll
    for (int j = 0; j < 8; ++j) l[j] = l[j] * scale[j] + rs[j];
#pragma unroll
    for (int t = 0; t < 4; ++t)
#pragma unroll
      for (int j = 0; j < 8; ++j) acc_o[t][j] *= scale[j];

    // stage P: C layout -> A-fragment slots (per-wave private region)
#pragma unroll
    for (int t = 0; t < 4; ++t)
#pragma unroll
      for (int j = 0; j < 8; ++j) {
        int col = t * 16 + lr;       // this lane's key column
        int row = j + 8 * half;      // this lane's query row (0..15)
        int ks = col >> 5, rem = col & 31, ah, fi;
        a_frag_slot(rem, ah, fi);
        ((__bf16*)&Ps[wave][ks][ah * 16 + row])[fi] = (__bf16)s[t][j];
      }

    // O += P @ V
#pragma unroll
    for (int ks = 0; ks < 2; ++ks) {
      v16bf pa = Ps[wave][ks][lane];
#pragma unroll
      for (int t = 0; t < 4; ++t)
        acc_o[t] = WMMA_BF16(pa, Vf[t][ks][lane], acc_o[t]);
    }
    __syncthreads();
  }

#pragma unroll
  for (int j = 0; j < 8; ++j) l[j] = 1.0f / l[j];
  const int row0 = blockIdx.x * 128 + wave * 16;
#pragma unroll
  for (int t = 0; t < 4; ++t)
#pragma unroll
    for (int j = 0; j < 8; ++j) {
      int r = row0 + j + 8 * half;
      O[(size_t)r * 64 + t * 16 + lr] = acc_o[t][j] * l[j];
    }
}

// ---------------------------------------------------------------------------
// logits = A @ B^T, A,B: [n,64].  128x128 block tile, K=64, write-BW bound.
// ---------------------------------------------------------------------------
__global__ __launch_bounds__(256) void outer_gemm(const float* __restrict__ A,
                                                  const float* __restrict__ B,
                                                  float* __restrict__ C, int n) {
  __shared__ v16bf Af[8][2][32];  // [wave][kstep][lane]
  __shared__ v16bf Bf[8][2][32];  // [ntile][kstep][lane]
  const int wave = threadIdx.x >> 5, lane = threadIdx.x & 31;
  const int half = lane >> 4, lr = lane & 15;
  const int row0 = blockIdx.y * 128, col0 = blockIdx.x * 128;
  for (int id = threadIdx.x; id < 128 * 64; id += 256) {
    int r = id >> 6, c = id & 63;
    float av = A[(size_t)(row0 + r) * 64 + c];
    float bv = B[(size_t)(col0 + r) * 64 + c];
    int ks = c >> 5, rem = c & 31;
    {  // A fragment slot
      int ah, fi; a_frag_slot(rem, ah, fi);
      ((__bf16*)&Af[r >> 4][ks][ah * 16 + (r & 15)])[fi] = (__bf16)av;
    }
    {  // B^T fragment slot: col=r, kdim=c
      int bh, fi; b_frag_slot(rem, bh, fi);
      ((__bf16*)&Bf[r >> 4][ks][bh * 16 + (r & 15)])[fi] = (__bf16)bv;
    }
  }
  __syncthreads();
  v8f acc[8] = {};
#pragma unroll
  for (int ks = 0; ks < 2; ++ks) {
    v16bf a = Af[wave][ks][lane];
#pragma unroll
    for (int t = 0; t < 8; ++t)
      acc[t] = WMMA_BF16(a, Bf[t][ks][lane], acc[t]);
  }
#pragma unroll
  for (int t = 0; t < 8; ++t)
#pragma unroll
    for (int j = 0; j < 8; ++j) {
      int r = row0 + wave * 16 + j + 8 * half;
      int c = col0 + t * 16 + lr;
      C[(size_t)r * n + c] = acc[t][j];
    }
}

// ---------------------------------------------------------------------------
// Host launch
// ---------------------------------------------------------------------------
extern "C" void kernel_launch(void* const* d_in, const int* in_sizes, int n_in,
                              void* d_out, int out_size, void* d_ws, size_t ws_size,
                              hipStream_t stream) {
  const int* g1 = (const int*)d_in[0];
  const int* g2 = (const int*)d_in[1];
  const float* x1 = (const float*)d_in[2];
  const float* x2 = (const float*)d_in[3];
  const float* W11 = (const float*)d_in[4];  const float* b11 = (const float*)d_in[5];
  const float* W12 = (const float*)d_in[6];  const float* b12 = (const float*)d_in[7];
  const float* W21 = (const float*)d_in[8];  const float* b21 = (const float*)d_in[9];
  const float* W22 = (const float*)d_in[10]; const float* b22 = (const float*)d_in[11];
  const float* q1 = (const float*)d_in[12];  const float* k1 = (const float*)d_in[13];
  const float* v1 = (const float*)d_in[14];
  const float* q2 = (const float*)d_in[15];  const float* k2 = (const float*)d_in[16];
  const float* v2 = (const float*)d_in[17];
  float* out = (float*)d_out;

  const int E_ = in_sizes[0] / 2;
  const int N_ = in_sizes[2] / 128;   // 8192

  // workspace carve-up (floats)
  float* w = (float*)d_ws;
  float* dinv1 = w; w += N_;
  float* dinv2 = w; w += N_;
  float* t256  = w; w += (size_t)N_ * 256;   // pre-agg hidden (layer1)
  float* hid   = w; w += (size_t)N_ * 256;   // post-agg hidden
  float* t64   = w; w += (size_t)N_ * 64;    // pre-agg (layer2)
  float* h1o   = w; w += (size_t)N_ * 64;
  float* h2o   = w; w += (size_t)N_ * 64;
  float* Qb    = w; w += (size_t)N_ * 64;
  float* Kb    = w; w += (size_t)N_ * 64;
  float* Vb    = w; w += (size_t)N_ * 64;
  float* o1    = w; w += (size_t)N_ * 64;
  float* o2    = w; w += (size_t)N_ * 64;

  const int T = 256;
  // degrees / dinv for both graphs
  gc_fill_ones<<<(N_ + T - 1) / T, T, 0, stream>>>(dinv1, N_);
  gc_deg_scatter<<<(E_ + T - 1) / T, T, 0, stream>>>(g1 + E_, E_, dinv1);
  gc_finish_dinv<<<(N_ + T - 1) / T, T, 0, stream>>>(dinv1, N_);
  gc_fill_ones<<<(N_ + T - 1) / T, T, 0, stream>>>(dinv2, N_);
  gc_deg_scatter<<<(E_ + T - 1) / T, T, 0, stream>>>(g2 + E_, E_, dinv2);
  gc_finish_dinv<<<(N_ + T - 1) / T, T, 0, stream>>>(dinv2, N_);

  // ---- GCN branch 1 ----
  gemm_bf16<<<dim3(N_ / 128, 256 / 64), T, 0, stream>>>(x1, W11, t256, N_, 256, 128);
  gc_agg_init<<<(N_ * 256 + T - 1) / T, T, 0, stream>>>(t256, dinv1, b11, hid, N_, 256);
  gc_agg_scatter<<<((size_t)E_ * 64 + T - 1) / T, T, 0, stream>>>(g1, g1 + E_, t256, dinv1, hid, E_, 256);
  gc_relu_ip<<<(N_ * 256 + T - 1) / T, T, 0, stream>>>(hid, N_ * 256);
  gemm_bf16<<<dim3(N_ / 128, 1), T, 0, stream>>>(hid, W12, t64, N_, 64, 256);
  gc_agg_init<<<(N_ * 64 + T - 1) / T, T, 0, stream>>>(t64, dinv1, b12, h1o, N_, 64);
  gc_agg_scatter<<<((size_t)E_ * 16 + T - 1) / T, T, 0, stream>>>(g1, g1 + E_, t64, dinv1, h1o, E_, 64);

  // ---- GCN branch 2 ----
  gemm_bf16<<<dim3(N_ / 128, 256 / 64), T, 0, stream>>>(x2, W21, t256, N_, 256, 128);
  gc_agg_init<<<(N_ * 256 + T - 1) / T, T, 0, stream>>>(t256, dinv2, b21, hid, N_, 256);
  gc_agg_scatter<<<((size_t)E_ * 64 + T - 1) / T, T, 0, stream>>>(g2, g2 + E_, t256, dinv2, hid, E_, 256);
  gc_relu_ip<<<(N_ * 256 + T - 1) / T, T, 0, stream>>>(hid, N_ * 256);
  gemm_bf16<<<dim3(N_ / 128, 1), T, 0, stream>>>(hid, W22, t64, N_, 64, 256);
  gc_agg_init<<<(N_ * 64 + T - 1) / T, T, 0, stream>>>(t64, dinv2, b22, h2o, N_, 64);
  gc_agg_scatter<<<((size_t)E_ * 16 + T - 1) / T, T, 0, stream>>>(g2, g2 + E_, t64, dinv2, h2o, E_, 64);

  // ---- cross attention 1: o1 = attn(h1@q1, h2@k1, h1@v1) ----
  gemm_bf16<<<dim3(N_ / 128, 1), T, 0, stream>>>(h1o, q1, Qb, N_, 64, 64);
  gemm_bf16<<<dim3(N_ / 128, 1), T, 0, stream>>>(h2o, k1, Kb, N_, 64, 64);
  gemm_bf16<<<dim3(N_ / 128, 1), T, 0, stream>>>(h1o, v1, Vb, N_, 64, 64);
  flash_attn<<<N_ / 128, T, 0, stream>>>(Qb, Kb, Vb, o1, N_);

  // ---- cross attention 2: o2 = attn(h2@q2, o1@k2, h2@v2) ----
  gemm_bf16<<<dim3(N_ / 128, 1), T, 0, stream>>>(h2o, q2, Qb, N_, 64, 64);
  gemm_bf16<<<dim3(N_ / 128, 1), T, 0, stream>>>(o1, k2, Kb, N_, 64, 64);
  gemm_bf16<<<dim3(N_ / 128, 1), T, 0, stream>>>(h2o, v2, Vb, N_, 64, 64);
  flash_attn<<<N_ / 128, T, 0, stream>>>(Qb, Kb, Vb, o2, N_);

  // ---- logits = o1 @ o2^T  (256 MB write, BW-bound) ----
  outer_gemm<<<dim3(N_ / 128, N_ / 128), T, 0, stream>>>(o1, o2, out, N_);
}
